// VectorQuantizer_68083821576369
// MI455X (gfx1250) — matrix operand order repaired and checked by reading, not compiled, not run on hardware
//
#include <hip/hip_runtime.h>

typedef __attribute__((ext_vector_type(16))) _Float16 v16h;
typedef __attribute__((ext_vector_type(8)))  _Float16 v8h;
typedef __attribute__((ext_vector_type(8)))  float    v8f;

#define BB      32
#define CCH     64
#define HW      4096
#define NROWS   (BB * HW)          // 131072 vectors
#define KCODES  1024
#define PITCH   72                 // f16 elems per LDS codebook row (bank-conflict-free b128)
#define NBLK    512
#define WPB     8                  // waves per block (256 threads, wave32)
#define NTILES  (NROWS / 16)       // 8192 row tiles
#define ZQ_ELEMS (NROWS * CCH)     // 8388608
#define LOSS_OFF ((size_t)ZQ_ELEMS)
#define IDX_OFF  ((size_t)ZQ_ELEMS + 1)
#define NPART   (NBLK * WPB)       // 4096 loss partials

#define SM_CB_BYTES  (KCODES * PITCH * 2)            // 147456
#define SM_NORM_OFF  SM_CB_BYTES
#define SM_IDX_OFF   (SM_NORM_OFF + KCODES * 4)      // 151552
#define SM_BYTES     (SM_IDX_OFF + WPB * 16 * 4)     // 152064

__global__ __launch_bounds__(256) void vq_main(const float* __restrict__ z,
                                               const float* __restrict__ cb,
                                               float* __restrict__ out,
                                               float* __restrict__ part) {
  extern __shared__ char smem[];
  _Float16* sCB  = (_Float16*)smem;
  float*    sNrm = (float*)(smem + SM_NORM_OFF);
  int*      sIdx = (int*)(smem + SM_IDX_OFF);

  // ---- stage codebook f32 -> f16 into LDS, compute ||e_k||^2 in f32 ----
  for (int r = threadIdx.x; r < KCODES; r += blockDim.x) {
    const float* src = cb + (size_t)r * CCH;
    _Float16*    dst = sCB + (size_t)r * PITCH;
    float nrm = 0.f;
#pragma unroll
    for (int c = 0; c < CCH; ++c) {
      float v = src[c];
      nrm = fmaf(v, v, nrm);
      dst[c] = (_Float16)v;
    }
    sNrm[r] = nrm;
  }
  __syncthreads();

  const int lane = threadIdx.x & 31;
  const int wave = threadIdx.x >> 5;
  const int half = lane >> 4;   // 0: lanes 0-15, 1: lanes 16-31
  const int lm   = lane & 15;

  float lsum = 0.f;

  for (int tile = blockIdx.x * WPB + wave; tile < NTILES; tile += NBLK * WPB) {
    const int row0 = tile << 4;            // 16 consecutive flat rows (same batch b)
    const int b    = row0 >> 12;           // n = b*4096 + hw
    const int hw0  = row0 & (HW - 1);
    const float* zb = z + (size_t)b * (CCH * HW) + (size_t)(hw0 + lm);

    // ---- A fragments (16-bit A 16x32 layout): lane m holds K 0-7,16-23; lane m+16: K 8-15,24-31
    v16h a0, a1;
#pragma unroll
    for (int e = 0; e < 16; ++e) {
      const int cbase = half * 8 + ((e < 8) ? e : (e + 8));
      a0[e] = (_Float16)zb[(size_t)cbase * HW];
      a1[e] = (_Float16)zb[(size_t)(cbase + 32) * HW];
    }

    float best[8];
    int   bidx[8];
#pragma unroll
    for (int r = 0; r < 8; ++r) { best[r] = 3.4e38f; bidx[r] = 0; }

    // ---- sweep all 1024 codes, 16 at a time; score = ||e||^2 - 2 z.e ----
#pragma unroll 2
    for (int nt = 0; nt < KCODES / 16; ++nt) {
      const int code = nt * 16 + lm;       // B column owned by this lane
      const _Float16* brow = sCB + (size_t)code * PITCH + half * 8;
      v8h l0 = *(const v8h*)(brow);        // K  0-7  (+8 for upper half-wave)
      v8h h0 = *(const v8h*)(brow + 16);   // K 16-23
      v8h l1 = *(const v8h*)(brow + 32);   // K 32-39
      v8h h1 = *(const v8h*)(brow + 48);   // K 48-55
      v16h b0, b1;
#pragma unroll
      for (int e = 0; e < 8; ++e) {
        b0[e] = l0[e]; b0[e + 8] = h0[e];
        b1[e] = l1[e]; b1[e + 8] = h1[e];
      }
      v8f acc = {0.f, 0.f, 0.f, 0.f, 0.f, 0.f, 0.f, 0.f};
      acc = __builtin_amdgcn_wmma_f32_16x16x32_f16(false, a0, false, b0, (short)0, acc, false, false);
      acc = __builtin_amdgcn_wmma_f32_16x16x32_f16(false, a1, false, b1, (short)0, acc, false, false);

      const float nrm = sNrm[code];
#pragma unroll
      for (int r = 0; r < 8; ++r) {
        const float s = fmaf(-2.f, acc[r], nrm);
        if (s < best[r]) { best[r] = s; bidx[r] = code; }
      }
    }

    // ---- argmin across the 16 code-classes (stays inside each 16-lane half) ----
#pragma unroll
    for (int m = 8; m >= 1; m >>= 1) {
#pragma unroll
      for (int r = 0; r < 8; ++r) {
        const float os = __shfl_xor(best[r], m, 32);
        const int   oi = __shfl_xor(bidx[r], m, 32);
        if (os < best[r] || (os == best[r] && oi < bidx[r])) { best[r] = os; bidx[r] = oi; }
      }
    }

    // D layout: lanes 0-15 slot r -> row r; lanes 16-31 slot r -> row 8+r
    if (lm == 0) {
#pragma unroll
      for (int r = 0; r < 8; ++r) {
        const int row = half * 8 + r;
        sIdx[wave * 16 + row] = bidx[r];
        out[IDX_OFF + (size_t)(row0 + row)] = (float)bidx[r];
      }
    }
    const int myIdx = sIdx[wave * 16 + lm];   // same-wave DS is in-order

    // ---- gather z_q (fp32 codebook), write (b,c,h,w), accumulate 1.25*MSE ----
    const float* qrow = cb + (size_t)myIdx * CCH;
    float*       ob   = out + (size_t)b * (CCH * HW) + (size_t)(hw0 + lm);
#pragma unroll
    for (int i = 0; i < 32; ++i) {
      const int c = 2 * i + half;           // lanes 0-15: even c, 16-31: odd c -> coalesced
      const float q = qrow[c];
      const float d = q - zb[(size_t)c * HW];
      lsum = fmaf(d, d, lsum);
      ob[(size_t)c * HW] = q;
    }
  }

  // ---- per-wave loss partial (deterministic two-pass) ----
#pragma unroll
  for (int m = 16; m >= 1; m >>= 1) lsum += __shfl_xor(lsum, m, 32);
  if (lane == 0) part[blockIdx.x * WPB + wave] = lsum;
}

__global__ __launch_bounds__(256) void vq_loss_reduce(const float* __restrict__ part,
                                                      float* __restrict__ out) {
  __shared__ float s[256];
  float v = 0.f;
  for (int i = threadIdx.x; i < NPART; i += 256) v += part[i];
  s[threadIdx.x] = v;
  __syncthreads();
  for (int st = 128; st > 0; st >>= 1) {
    if ((int)threadIdx.x < st) s[threadIdx.x] += s[threadIdx.x + st];
    __syncthreads();
  }
  if (threadIdx.x == 0) out[LOSS_OFF] = s[0] * (1.25f / (float)ZQ_ELEMS);
}

extern "C" void kernel_launch(void* const* d_in, const int* in_sizes, int n_in,
                              void* d_out, int out_size, void* d_ws, size_t ws_size,
                              hipStream_t stream) {
  (void)in_sizes; (void)n_in; (void)out_size; (void)ws_size;
  const float* z  = (const float*)d_in[0];
  const float* cb = (const float*)d_in[1];
  float* out  = (float*)d_out;
  float* part = (float*)d_ws;   // 4096 floats of scratch

  hipFuncSetAttribute(reinterpret_cast<const void*>(vq_main),
                      hipFuncAttributeMaxDynamicSharedMemorySize, SM_BYTES);
  vq_main<<<NBLK, 256, SM_BYTES, stream>>>(z, cb, out, part);
  vq_loss_reduce<<<1, 256, 0, stream>>>(part, out);
}
